// TransformerBlock_63840393888551
// MI455X (gfx1250) — matrix operand order repaired and checked by reading, not compiled, not run on hardware
//
#include <hip/hip_runtime.h>
#include <hip/hip_bf16.h>
#include <cmath>

// ---- fixed problem dims ----
#define D_    1024
#define S_    2048
#define H_    16
#define DH_   64
#define WIN_  512
#define ROWS_ 4096   // B*S

typedef __attribute__((ext_vector_type(16))) _Float16 v16h;
typedef __attribute__((ext_vector_type(8)))  _Float16 v8h;
typedef __attribute__((ext_vector_type(4)))  _Float16 v4h;
typedef __attribute__((ext_vector_type(2)))  _Float16 v2h;
typedef __attribute__((ext_vector_type(8)))  float    v8f;
typedef __attribute__((ext_vector_type(4)))  unsigned v4u;
typedef __attribute__((ext_vector_type(8)))  int      v8i;
typedef __attribute__((ext_vector_type(4)))  int      v4i;

// LDS row stride for TDM-staged tiles: 64B data + 16B pad = 80B = 40 halves.
#define LROW_ 40

// Build a 16-bit A-fragment (16x32 MxK) from two contiguous 8-half chunks.
static __device__ __forceinline__ v16h make_afrag(const _Float16* lo,
                                                  const _Float16* hi) {
  v8h l = *(const v8h*)lo;
  v8h h = *(const v8h*)hi;
  v16h a;
#pragma unroll
  for (int t = 0; t < 8; ++t) { a[t] = l[t]; a[8 + t] = h[t]; }
  return a;
}

// ---------------------------------------------------------------------------
// Tensor Data Mover: 2D tile load Global->LDS (ISA 08_async_tensor §8).
// data_size = 8B units; rows of `row8` units with stride `stride8` units.
// LDS padding: after every 16 dwords (64B row) insert 4 dwords (16B) so the
// LDS row stride is 80B (16B-aligned ds_load_b128, conflict-free banks).
// ---------------------------------------------------------------------------
static __device__ __forceinline__ void tdm_load2d(const _Float16* gsrc,
                                                  unsigned lds_byte_off,
                                                  unsigned rows, unsigned row8,
                                                  unsigned long long stride8) {
  const unsigned long long ga = (unsigned long long)(size_t)gsrc;
  v4u g0;
  g0[0] = 1u;                                   // count=1, user descriptor
  g0[1] = lds_byte_off;                         // LDS dest (bytes)
  g0[2] = (unsigned)ga;                         // global addr [31:0]
  g0[3] = (unsigned)((ga >> 32) & 0x1ffffffu)   // global addr [56:32]
          | (2u << 30);                         // type = 2 ("image")
  // flags: data_size=3 (8B), pad_enable, pad_interval=3 (16dw), pad_amount=3 (4dw)
  const unsigned flags = (3u << 16) | (1u << 20) | (3u << 22) | (3u << 25);
  const unsigned long long td0 = stride8;       // tensor_dim0 = stride: no OOB
  const unsigned td1 = rows;
  v8i g1;
  g1[0] = (int)flags;
  g1[1] = (int)((td0 & 0xffffu) << 16);                         // abar=0 | td0.lo
  g1[2] = (int)(((td0 >> 16) & 0xffffu) | ((td1 & 0xffffu) << 16));
  g1[3] = (int)(((td1 >> 16) & 0xffffu) | ((row8 & 0xffffu) << 16)); // tile_dim0
  g1[4] = (int)(rows & 0xffffu);                                // tile_dim1, dim2=0
  g1[5] = (int)(td0 & 0xffffffffu);                             // stride0 [31:0]
  g1[6] = (int)((td0 >> 32) & 0xffffu);                         // stride0 hi, stride1=0
  g1[7] = 0;
  const v4i z4 = {0, 0, 0, 0};                  // 2D tile: groups 2/3 unused
  const v8i z8 = {0, 0, 0, 0, 0, 0, 0, 0};
  __builtin_amdgcn_tensor_load_to_lds(g0, g1, z4, z4, z8, 0);
}

// ---------------------------------------------------------------------------
// LayerNorm: one block (256 thr) per row of 1024. Writes f16 (always) and
// optionally f32.
// ---------------------------------------------------------------------------
__global__ __launch_bounds__(256) void ln_k(const float* __restrict__ x,
                                            const float* __restrict__ g,
                                            const float* __restrict__ beta,
                                            float* __restrict__ of32,
                                            _Float16* __restrict__ of16) {
  __shared__ float red[16];
  const int row = blockIdx.x;
  const int t = threadIdx.x;
  const float4 v = ((const float4*)(x + (size_t)row * D_))[t];
  float s  = v.x + v.y + v.z + v.w;
  float s2 = v.x * v.x + v.y * v.y + v.z * v.z + v.w * v.w;
#pragma unroll
  for (int m = 16; m >= 1; m >>= 1) {
    s  += __shfl_xor(s,  m, 32);
    s2 += __shfl_xor(s2, m, 32);
  }
  if ((t & 31) == 0) { red[t >> 5] = s; red[8 + (t >> 5)] = s2; }
  __syncthreads();
  float S = 0.f, S2 = 0.f;
#pragma unroll
  for (int i = 0; i < 8; ++i) { S += red[i]; S2 += red[8 + i]; }
  const float mean = S * (1.0f / D_);
  const float rstd = rsqrtf(S2 * (1.0f / D_) - mean * mean + 1e-5f);
  const float4 gv = ((const float4*)g)[t];
  const float4 bv = ((const float4*)beta)[t];
  const float y0 = (v.x - mean) * rstd * gv.x + bv.x;
  const float y1 = (v.y - mean) * rstd * gv.y + bv.y;
  const float y2 = (v.z - mean) * rstd * gv.z + bv.z;
  const float y3 = (v.w - mean) * rstd * gv.w + bv.w;
  if (of32) ((float4*)(of32 + (size_t)row * D_))[t] = make_float4(y0, y1, y2, y3);
  v4h hv;
  hv[0] = (_Float16)y0; hv[1] = (_Float16)y1;
  hv[2] = (_Float16)y2; hv[3] = (_Float16)y3;
  ((v4h*)(of16 + (size_t)row * D_))[t] = hv;
}

// ---------------------------------------------------------------------------
// Weight convert+transpose: fp32 [K,N] -> f16 [N,K] via LDS 32x32 tiles.
// ---------------------------------------------------------------------------
__global__ __launch_bounds__(256) void wt_k(const float* __restrict__ W,
                                            _Float16* __restrict__ WT,
                                            int K, int N) {
  __shared__ _Float16 tile[32][33];
  const int n0 = blockIdx.x * 32, k0 = blockIdx.y * 32;
  const int tx = threadIdx.x & 31, ty = threadIdx.x >> 5;
#pragma unroll
  for (int i = 0; i < 4; ++i)
    tile[ty + 8 * i][tx] = (_Float16)W[(size_t)(k0 + ty + 8 * i) * N + n0 + tx];
  __syncthreads();
#pragma unroll
  for (int i = 0; i < 4; ++i)
    WT[(size_t)(n0 + ty + 8 * i) * K + k0 + tx] = tile[tx][ty + 8 * i];
}

// ---------------------------------------------------------------------------
// WMMA GEMM with TDM double-buffered LDS staging.
// C[M,N] = A[M,K](f16,row) * BT[N,K](f16,row)^T + bias, fused epilogue.
// One wave -> 16x64 tile; 8 waves/block -> 128x64 block tile.
// Per k-step (32): TDM loads next A (128x64B) and BT (64x64B) tiles into LDS
// (80B padded rows) while WMMAs consume the current buffer.
// mode 0: of32 = acc + bias (+ res)           (fp32 out)
// mode 1: f16 out scattered to [B,H,S,DH]     (Q,K)
// mode 2: f16 out scattered to [B,H,DH,S]     (V transposed)
// mode 3: f16 out = gelu(acc + bias)          (FFN hidden)
// ---------------------------------------------------------------------------
__global__ __launch_bounds__(256) void gemm_k(const _Float16* __restrict__ A,
                                              const _Float16* __restrict__ BT,
                                              const float* __restrict__ bias,
                                              const float* __restrict__ res,
                                              float* __restrict__ of32,
                                              _Float16* __restrict__ of16,
                                              int M, int N, int K, int mode) {
  __shared__ _Float16 smA[2][128 * LROW_];
  __shared__ _Float16 smB[2][64 * LROW_];

  const int lane = threadIdx.x & 31, wid = threadIdx.x >> 5;
  const int hf = lane >> 4, lm = lane & 15;
  const int mrow0 = blockIdx.y * 128;
  const int nbase = blockIdx.x * 64;
  const unsigned long long s8 = (unsigned long long)(K >> 2);  // row stride, 8B units

  // prologue: stage k0=0 tiles
  if (wid == 0) {
    tdm_load2d(A  + (size_t)mrow0 * K, (unsigned)(size_t)&smA[0][0], 128, 8, s8);
    tdm_load2d(BT + (size_t)nbase * K, (unsigned)(size_t)&smB[0][0], 64, 8, s8);
    __builtin_amdgcn_s_wait_tensorcnt(0);
  }
  __syncthreads();

  v8f acc[4] = {};
  int buf = 0;
  for (int k0 = 0; k0 < K; k0 += 32) {
    // kick off next tiles into the other buffer
    if (wid == 0 && (k0 + 32) < K) {
      tdm_load2d(A + (size_t)mrow0 * K + (k0 + 32),
                 (unsigned)(size_t)&smA[buf ^ 1][0], 128, 8, s8);
      tdm_load2d(BT + (size_t)nbase * K + (k0 + 32),
                 (unsigned)(size_t)&smB[buf ^ 1][0], 64, 8, s8);
    }
    // compute on current buffer
    const _Float16* ar = &smA[buf][(wid * 16 + lm) * LROW_];
    const v16h a = make_afrag(ar + hf * 16, ar + hf * 16 + 8);
#pragma unroll
    for (int t = 0; t < 4; ++t) {
      const v16h b = *(const v16h*)(&smB[buf][(t * 16 + lm) * LROW_ + hf * 16]);
      acc[t] = __builtin_amdgcn_wmma_f32_16x16x32_f16(
          false, a, false, b, (short)0, acc[t], false, false);
    }
    if (wid == 0) __builtin_amdgcn_s_wait_tensorcnt(0);
    __syncthreads();
    buf ^= 1;
  }

#pragma unroll
  for (int t = 0; t < 4; ++t) {
    const int col = nbase + t * 16 + lm;
    const float bv = bias ? bias[col] : 0.f;
#pragma unroll
    for (int r = 0; r < 8; ++r) {
      const int row = mrow0 + wid * 16 + r + hf * 8;
      const float v = acc[t][r] + bv;
      if (mode == 0) {
        const size_t o = (size_t)row * N + col;
        of32[o] = v + (res ? res[o] : 0.f);
      } else if (mode == 3) {
        const float gl = 0.5f * v * (1.f + erff(v * 0.70710678118654752f));
        of16[(size_t)row * N + col] = (_Float16)gl;
      } else {
        const int hh = col >> 6, dh = col & 63;
        const int bb = row >> 11, ss = row & (S_ - 1);
        const size_t o =
            (mode == 1) ? ((size_t)((bb * H_ + hh) * S_ + ss) * DH_ + dh)
                        : ((size_t)((bb * H_ + hh) * DH_ + dh) * S_ + ss);
        of16[o] = (_Float16)v;
      }
    }
  }
}

// ---------------------------------------------------------------------------
// Flash attention, sliding window. One wave per (b,h,16-query tile).
// Computes S^T = K*Q^T (softmax key-reduction stays in-register), online
// softmax, register-shuffle transpose of P, ctx^T = Vt*P^T.
// q,k: f16 [B,H,S,DH]; vt: f16 [B,H,DH,S]; ctx out: f16 [B*S, D].
// ---------------------------------------------------------------------------
__global__ __launch_bounds__(256) void attn_k(const _Float16* __restrict__ q,
                                              const _Float16* __restrict__ k,
                                              const _Float16* __restrict__ vt,
                                              _Float16* __restrict__ ctx) {
  const int lane = threadIdx.x & 31, wid = threadIdx.x >> 5;
  const int hf = lane >> 4, ln = lane & 15;
  const int w = blockIdx.x * 8 + wid;        // 0..4095
  const int qb = w & 127;
  const int hh = (w >> 7) & 15;
  const int bb = w >> 11;
  const int i0 = qb * 16;

  const _Float16* qp = q  + (size_t)(bb * H_ + hh) * S_ * DH_;
  const _Float16* kp = k  + (size_t)(bb * H_ + hh) * S_ * DH_;
  const _Float16* vp = vt + (size_t)(bb * H_ + hh) * DH_ * S_;

  // Q^T B-fragments: lane = query column; K-dim = dh (2 chunks of 32).
  v16h bq[2];
#pragma unroll
  for (int c = 0; c < 2; ++c)
    bq[c] = *(const v16h*)(qp + (size_t)(i0 + ln) * DH_ + c * 32 + hf * 16);

  v8f cacc[4] = {};
  float mrun = -3.0e38f, lrun = 0.f;
  const int i = i0 + ln;  // this lane's query index
  const int lo = i0 - WIN_;
  const int jstart = lo > 0 ? (lo & ~31) : 0;

  for (int j0 = jstart; j0 <= i0 + 15; j0 += 32) {
    // ---- S^T tiles: st[tb] covers keys j0+tb*16 .. +15 (M), queries (N) ----
    v8f st[2] = {};
#pragma unroll
    for (int tb = 0; tb < 2; ++tb) {
      const _Float16* kr = kp + (size_t)(j0 + tb * 16 + ln) * DH_;
#pragma unroll
      for (int c = 0; c < 2; ++c) {
        const v16h a = make_afrag(kr + c * 32 + hf * 16, kr + c * 32 + hf * 16 + 8);
        st[tb] = __builtin_amdgcn_wmma_f32_16x16x32_f16(
            false, a, false, bq[c], (short)0, st[tb], false, false);
      }
    }
    // ---- mask + online softmax (per lane = one query) ----
    float p[16];
    float rmax = -3.0e38f;
#pragma unroll
    for (int tb = 0; tb < 2; ++tb) {
#pragma unroll
      for (int r = 0; r < 8; ++r) {
        const int j = j0 + tb * 16 + r + hf * 8;
        float s = st[tb][r] * 0.125f;  // 1/sqrt(64)
        const bool blocked = (j > i) || (j < i - WIN_);
        s = blocked ? -3.0e38f : s;
        p[tb * 8 + r] = s;
        rmax = fmaxf(rmax, s);
      }
    }
    rmax = fmaxf(rmax, __shfl_xor(rmax, 16, 32));
    const float mnew = fmaxf(mrun, rmax);
    const float scale = __expf(mrun - mnew);
    float lsum = 0.f;
#pragma unroll
    for (int e = 0; e < 16; ++e) {
      const float pe = (p[e] > -1.0e37f) ? __expf(p[e] - mnew) : 0.f;
      p[e] = pe;
      lsum += pe;
    }
    lsum += __shfl_xor(lsum, 16, 32);
    lrun = lrun * scale + lsum;
    mrun = mnew;
#pragma unroll
    for (int t = 0; t < 4; ++t)
#pragma unroll
      for (int r = 0; r < 8; ++r) cacc[t][r] *= scale;

    // ---- register transpose of P into a B-fragment (K=j, col=query) ----
    v16h bp;
#pragma unroll
    for (int r = 0; r < 8; ++r) {
      const float o0 = p[r];        // tile0, own half
      const float o1 = p[8 + r];    // tile1, own half
      const float c0 = __shfl_xor(o0, 16, 32);
      const float c1 = __shfl_xor(o1, 16, 32);
      bp[r]     = (hf == 0) ? (_Float16)o0 : (_Float16)c1;
      bp[8 + r] = (hf == 0) ? (_Float16)c0 : (_Float16)o1;
    }

    // ---- ctx^T += Vt * P^T : 4 dh tiles of 16 ----
#pragma unroll
    for (int t = 0; t < 4; ++t) {
      const _Float16* vr = vp + (size_t)(t * 16 + ln) * S_ + j0;
      const v16h a = make_afrag(vr + hf * 16, vr + hf * 16 + 8);
      cacc[t] = __builtin_amdgcn_wmma_f32_16x16x32_f16(
          false, a, false, bp, (short)0, cacc[t], false, false);
    }
  }

  // ---- normalize + store ctx (f16, row-major [B*S, D]) ----
  const float inv = 1.f / lrun;
  _Float16* cp = ctx + ((size_t)(bb * S_ + i0 + ln)) * D_ + hh * DH_;
#pragma unroll
  for (int t = 0; t < 4; ++t) {
#pragma unroll
    for (int rp = 0; rp < 4; ++rp) {
      const int dh = t * 16 + 2 * rp + hf * 8;
      v2h pr;
      pr[0] = (_Float16)(cacc[t][2 * rp] * inv);
      pr[1] = (_Float16)(cacc[t][2 * rp + 1] * inv);
      *(v2h*)(cp + dh) = pr;
    }
  }
}

// ---------------------------------------------------------------------------
extern "C" void kernel_launch(void* const* d_in, const int* in_sizes, int n_in,
                              void* d_out, int out_size, void* d_ws,
                              size_t ws_size, hipStream_t stream) {
  (void)in_sizes; (void)n_in; (void)out_size; (void)ws_size;
  const float* x   = (const float*)d_in[0];
  const float* Wq  = (const float*)d_in[1];
  const float* bq  = (const float*)d_in[2];
  const float* Wk  = (const float*)d_in[3];
  const float* bk  = (const float*)d_in[4];
  const float* Wv  = (const float*)d_in[5];
  const float* bv  = (const float*)d_in[6];
  const float* Wo  = (const float*)d_in[7];
  const float* bo  = (const float*)d_in[8];
  const float* g1  = (const float*)d_in[9];
  const float* b1  = (const float*)d_in[10];
  const float* g2  = (const float*)d_in[11];
  const float* b2  = (const float*)d_in[12];
  const float* W1  = (const float*)d_in[13];
  const float* bf1 = (const float*)d_in[14];
  const float* W2  = (const float*)d_in[15];
  const float* bf2 = (const float*)d_in[16];

  char* ws = (char*)d_ws;
  size_t off = 0;
  auto alloc = [&](size_t bytes) -> char* {
    char* p = ws + off;
    off += (bytes + 255) & ~(size_t)255;
    return p;
  };

  _Float16* xn16 = (_Float16*)alloc((size_t)ROWS_ * D_ * 2);
  float*    xn32 = (float*)   alloc((size_t)ROWS_ * D_ * 4);
  _Float16* WqT  = (_Float16*)alloc((size_t)D_ * D_ * 2);
  _Float16* WkT  = (_Float16*)alloc((size_t)D_ * D_ * 2);
  _Float16* WvT  = (_Float16*)alloc((size_t)D_ * D_ * 2);
  _Float16* WoT  = (_Float16*)alloc((size_t)D_ * D_ * 2);
  _Float16* W1T  = (_Float16*)alloc((size_t)4 * D_ * D_ * 2);  // [4096,1024]
  _Float16* W2T  = (_Float16*)alloc((size_t)4 * D_ * D_ * 2);  // [1024,4096]
  _Float16* qf   = (_Float16*)alloc((size_t)ROWS_ * D_ * 2);
  _Float16* kf   = (_Float16*)alloc((size_t)ROWS_ * D_ * 2);
  _Float16* vtf  = (_Float16*)alloc((size_t)ROWS_ * D_ * 2);
  _Float16* ctxf = (_Float16*)alloc((size_t)ROWS_ * D_ * 2);
  float*    x1   = (float*)   alloc((size_t)ROWS_ * D_ * 4);
  _Float16* x1n  = (_Float16*)alloc((size_t)ROWS_ * D_ * 2);
  _Float16* h16  = qf;  // reuse q/k/vt/ctx region (32MB) for FFN hidden

  const dim3 blk(256);

  // 1) LN1 -> xn (f32 + f16)
  ln_k<<<ROWS_, blk, 0, stream>>>(x, g1, b1, xn32, xn16);

  // 2) weight convert+transpose to f16 [N,K]
  wt_k<<<dim3(32, 32),  blk, 0, stream>>>(Wq, WqT, D_, D_);
  wt_k<<<dim3(32, 32),  blk, 0, stream>>>(Wk, WkT, D_, D_);
  wt_k<<<dim3(32, 32),  blk, 0, stream>>>(Wv, WvT, D_, D_);
  wt_k<<<dim3(32, 32),  blk, 0, stream>>>(Wo, WoT, D_, D_);
  wt_k<<<dim3(128, 32), blk, 0, stream>>>(W1, W1T, D_, 4 * D_);
  wt_k<<<dim3(32, 128), blk, 0, stream>>>(W2, W2T, 4 * D_, D_);

  // 3) QKV projections (WMMA + TDM staging), scattered into head layouts
  gemm_k<<<dim3(16, 32), blk, 0, stream>>>(xn16, WqT, bq, nullptr, nullptr, qf,
                                           ROWS_, D_, D_, 1);
  gemm_k<<<dim3(16, 32), blk, 0, stream>>>(xn16, WkT, bk, nullptr, nullptr, kf,
                                           ROWS_, D_, D_, 1);
  gemm_k<<<dim3(16, 32), blk, 0, stream>>>(xn16, WvT, bv, nullptr, nullptr, vtf,
                                           ROWS_, D_, D_, 2);

  // 4) sliding-window flash attention (WMMA)
  attn_k<<<512, blk, 0, stream>>>(qf, kf, vtf, ctxf);

  // 5) output projection + residual (x1 = ctx@Wo + bo + xn)
  gemm_k<<<dim3(16, 32), blk, 0, stream>>>(ctxf, WoT, bo, xn32, x1, nullptr,
                                           ROWS_, D_, D_, 0);

  // 6) LN2 -> x1n (f16)
  ln_k<<<ROWS_, blk, 0, stream>>>(x1, g2, b2, nullptr, x1n);

  // 7) FFN: h = gelu(x1n@W1 + bf1)
  gemm_k<<<dim3(64, 32), blk, 0, stream>>>(x1n, W1T, bf1, nullptr, nullptr, h16,
                                           ROWS_, 4 * D_, D_, 3);

  // 8) out = h@W2 + bf2 + x1
  gemm_k<<<dim3(16, 32), blk, 0, stream>>>(h16, W2T, bf2, x1, (float*)d_out,
                                           nullptr, ROWS_, D_, 4 * D_, 0);
}
